// Seq2Space_49727131353112
// MI455X (gfx1250) — matrix operand order repaired and checked by reading, not compiled, not run on hardware
//
#include <hip/hip_runtime.h>

// Problem constants (match reference)
#define B_    8
#define L_    16384
#define CIN_  256
#define COUT_ 256
#define LS_   4096
#define D_    64

// ---------------------------------------------------------------------------
// Types: ext_vector types only (trivial -> safe in unions, b128 loads)
// ---------------------------------------------------------------------------
typedef __attribute__((ext_vector_type(16))) __bf16   v16bf;
typedef __attribute__((ext_vector_type(8)))  float    v8f;
typedef __attribute__((ext_vector_type(4)))  unsigned uint4v;
typedef __attribute__((ext_vector_type(4)))  float    float4v;

union Frag32 {            // one WMMA A/B fragment: 16 bf16 = 8 dwords = 32B
    uint4v   q[2];
    unsigned u[8];
    v16bf    v;
};

__device__ __forceinline__ unsigned f2bf_pk(float lo, float hi) {
    // round-to-nearest-even f32 -> bf16, packed pair
    unsigned a = __float_as_uint(lo);
    unsigned b = __float_as_uint(hi);
    a = (a + 0x7FFFu + ((a >> 16) & 1u)) >> 16;
    b = (b + 0x7FFFu + ((b >> 16) & 1u)) >> 16;
    return (a & 0xFFFFu) | (b << 16);
}

__device__ __forceinline__ v8f vzero8() {
    v8f z;
#pragma unroll
    for (int i = 0; i < 8; ++i) z[i] = 0.0f;
    return z;
}

__device__ __forceinline__ v8f wmma_bf16(const Frag32& a, const Frag32& b, v8f c) {
    return __builtin_amdgcn_wmma_f32_16x16x32_bf16(false, a.v, false, b.v,
                                                   (short)0, c, false, false);
}

// Shared-memory pool reused for A-tile double buffer and epilogue transpose.
// Both views are exactly 2560 dwords = 10240 bytes.
union SMem {
    unsigned a[2][64 * 20];      // bf16-pair A staging, padded stride 20 dwords
    float    ep[8][16][20];      // per-wave 16x16 f32 tile, padded stride 20
};

// Epilogue helper: D fragment (f32) -> per-wave LDS tile -> coalesced b128
// stores (2 per lane instead of 8 scattered b32). Wave-private region; DS ops
// from one wave are in-order so no barrier is needed.
__device__ __forceinline__ void tile_store(float* __restrict__ outp, long rbase,
                                           int colbase, const float* __restrict__ bias,
                                           v8f acc, float (*ep)[20], int lane) {
    const int n = lane & 15, hi = lane >> 4;
    const float bv = bias[colbase + n];
#pragma unroll
    for (int v = 0; v < 8; ++v) ep[v + 8 * hi][n] = acc[v] + bv;
    const int row = lane >> 1, half = lane & 1;
    const float4v w0 = *(const float4v*)&ep[row][half * 8];
    const float4v w1 = *(const float4v*)&ep[row][half * 8 + 4];
    float* o = outp + (rbase + row) * (long)COUT_ + colbase + half * 8;
    *(float4v*)o = w0;
    *(float4v*)(o + 4) = w1;
}

// ---------------------------------------------------------------------------
// K1: pack a [256][256] row-major f32 weight (K x N) into per-lane B-fragment
// order: dword index = ((kt*16 + nt)*32 + lane)*8 + v
// ---------------------------------------------------------------------------
__global__ __launch_bounds__(256) void pack_b_kernel(const float* __restrict__ W,
                                                     unsigned* __restrict__ out) {
    const int tile = blockIdx.x * 8 + (threadIdx.x >> 5);   // 0..127 = kt*16+nt
    const int lane = threadIdx.x & 31;
    const int kt = tile >> 4, nt = tile & 15;
    const int n = nt * 16 + (lane & 15);
    const int kbase = kt * 32 + ((lane >> 4) << 4);
    unsigned* dst = out + ((size_t)tile * 32 + lane) * 8;
#pragma unroll
    for (int v = 0; v < 8; ++v) {
        const int k = kbase + 2 * v;
        dst[v] = f2bf_pk(W[(size_t)k * COUT_ + n], W[(size_t)(k + 1) * COUT_ + n]);
    }
}

// K2: f32 -> bf16-pair row-major conversion (for Wc as A-side matrix)
__global__ void cvt_pairs_kernel(const float* __restrict__ in,
                                 unsigned* __restrict__ out, int ndw) {
    int i = blockIdx.x * blockDim.x + threadIdx.x;
    if (i < ndw) out[i] = f2bf_pk(in[2 * i], in[2 * i + 1]);
}

// ---------------------------------------------------------------------------
// K3: fused maxpool(4) + dual GEMM:  xa = maxpool(x)@Wa+ba, xb = ...@Wb+bb
// grid 512 blocks (64 pooled rows each), 256 threads = 8 waves.
// waves: mslice = wv&3 (16 rows each), mat = wv>>2 (0:Wa 1:Wb).
// Each wave: full N=256 -> 16 accumulator tiles, K loop 8 x 32.
// B fragments are preloaded in chunks of 8 so loads cluster ahead of WMMAs.
// ---------------------------------------------------------------------------
__global__ __launch_bounds__(256) void pool_dual_gemm(
        const float* __restrict__ x,
        const unsigned* __restrict__ WaPk, const unsigned* __restrict__ WbPk,
        const float* __restrict__ ba, const float* __restrict__ bb,
        float* __restrict__ xa, float* __restrict__ xb) {
    __shared__ SMem smem;
    const int tid = threadIdx.x;
    const int lane = tid & 31;
    const int wv = tid >> 5;
    const int mslice = wv & 3;
    const int mat = wv >> 2;
    const long gr0 = (long)blockIdx.x * 64;     // pooled row base in [0, B*LS)

    v8f acc[16];
#pragma unroll
    for (int i = 0; i < 16; ++i) acc[i] = vzero8();

    const unsigned* BPk = mat ? WbPk : WaPk;

    for (int kt = 0; kt < 8; ++kt) {
        unsigned* buf = smem.a[kt & 1];
        {   // stage pooled A tile: 64 rows x 32 cols, bf16
            const int r  = tid >> 2;
            const int c8 = (tid & 3) * 8;
            const float* src = x + ((gr0 + r) * 4) * (long)CIN_ + kt * 32 + c8;
            float4v m0 = *(const float4v*)(src);
            float4v m1 = *(const float4v*)(src + 4);
#pragma unroll
            for (int s = 1; s < 4; ++s) {
                float4v a0 = *(const float4v*)(src + (long)s * CIN_);
                float4v a1 = *(const float4v*)(src + (long)s * CIN_ + 4);
                m0.x = fmaxf(m0.x, a0.x); m0.y = fmaxf(m0.y, a0.y);
                m0.z = fmaxf(m0.z, a0.z); m0.w = fmaxf(m0.w, a0.w);
                m1.x = fmaxf(m1.x, a1.x); m1.y = fmaxf(m1.y, a1.y);
                m1.z = fmaxf(m1.z, a1.z); m1.w = fmaxf(m1.w, a1.w);
            }
            unsigned* dst = &buf[r * 20 + (c8 >> 1)];
            dst[0] = f2bf_pk(m0.x, m0.y); dst[1] = f2bf_pk(m0.z, m0.w);
            dst[2] = f2bf_pk(m1.x, m1.y); dst[3] = f2bf_pk(m1.z, m1.w);
            if (kt < 7) __builtin_prefetch(src + 32, 0, 3);   // next K tile
        }
        __syncthreads();

        Frag32 a;
        {   // A fragment: row m, dword cols cd..cd+7
            const int m  = mslice * 16 + (lane & 15);
            const int cd = (lane >> 4) * 8;
            a.q[0] = *(const uint4v*)&buf[m * 20 + cd];
            a.q[1] = *(const uint4v*)&buf[m * 20 + cd + 4];
        }
        const uint4v* bp = (const uint4v*)BPk + ((size_t)(kt * 16) * 32 + lane) * 2;
#pragma unroll
        for (int half = 0; half < 2; ++half) {
            Frag32 bfr[8];
#pragma unroll
            for (int t = 0; t < 8; ++t) {        // 16 clustered b128 loads
                bfr[t].q[0] = bp[0];
                bfr[t].q[1] = bp[1];
                bp += 64;
            }
#pragma unroll
            for (int t = 0; t < 8; ++t)
                acc[half * 8 + t] = wmma_bf16(a, bfr[t], acc[half * 8 + t]);
        }
    }

    // epilogue: bias + coalesced stores via per-wave LDS transpose
    __syncthreads();                              // staging buffers retired
    float* outp = mat ? xb : xa;
    const float* bias = mat ? bb : ba;
    const long rbase = gr0 + mslice * 16;
#pragma unroll
    for (int nt = 0; nt < 16; ++nt)
        tile_store(outp, rbase, nt * 16, bias, acc[nt], smem.ep[wv], lane);
}

// ---------------------------------------------------------------------------
// K4: partial max over l per (b,c). grid = B*16, block 256 (thread = channel)
// ---------------------------------------------------------------------------
__global__ __launch_bounds__(256) void reduce_max_kernel(const float* __restrict__ xa,
                                                         float* __restrict__ pmax) {
    const int b = blockIdx.x >> 4, chunk = blockIdx.x & 15, c = threadIdx.x;
    const float* p = xa + ((long)b * LS_ + chunk * 256) * COUT_ + c;
    float m = -3.402823466e+38f;
    for (int l = 0; l < 256; ++l) m = fmaxf(m, p[(long)l * COUT_]);
    pmax[(size_t)blockIdx.x * 256 + c] = m;
}

// K5: partial Z = sum exp, W = sum exp*(xb[l] - xb[(l+d)%LS]) with d = c%64
__global__ __launch_bounds__(256) void reduce_sums_kernel(
        const float* __restrict__ xa, const float* __restrict__ xb,
        const float* __restrict__ pmax,
        float* __restrict__ pZ, float* __restrict__ pW) {
    const int b = blockIdx.x >> 4, chunk = blockIdx.x & 15, c = threadIdx.x;
    float gm = -3.402823466e+38f;
#pragma unroll
    for (int j = 0; j < 16; ++j) gm = fmaxf(gm, pmax[((size_t)b * 16 + j) * 256 + c]);
    const int d = c & (D_ - 1);
    const long base = (long)b * LS_ * COUT_;
    float Z = 0.f, W = 0.f;
    for (int i = 0; i < 256; ++i) {
        const int l = chunk * 256 + i;
        const float e  = __expf(xa[base + (long)l * COUT_ + c] - gm);
        const float bv = xb[base + (long)l * COUT_ + c];
        const float bs = xb[base + (long)((l + d) & (LS_ - 1)) * COUT_ + c];
        Z += e;
        W += e * (bv - bs);
    }
    pZ[(size_t)blockIdx.x * 256 + c] = Z;
    pW[(size_t)blockIdx.x * 256 + c] = W;
}

// K6: combine -> s[b,c] = W/Z. grid B, block 256.
__global__ __launch_bounds__(256) void reduce_final_kernel(
        const float* __restrict__ pZ, const float* __restrict__ pW,
        float* __restrict__ s) {
    const int b = blockIdx.x, c = threadIdx.x;
    float Z = 0.f, W = 0.f;
#pragma unroll
    for (int j = 0; j < 16; ++j) {
        Z += pZ[((size_t)b * 16 + j) * 256 + c];
        W += pW[((size_t)b * 16 + j) * 256 + c];
    }
    s[(size_t)b * 256 + c] = W / Z;
}

// K7: bfused[b][j] = bo[j] + sum_c bc[c]*s[b,c]*Wo[(c%64)*4 + c/64][j]
__global__ __launch_bounds__(256) void bfused_kernel(
        const float* __restrict__ bc, const float* __restrict__ Wo,
        const float* __restrict__ bo, const float* __restrict__ s,
        float* __restrict__ bf) {
    const int b = blockIdx.x, j = threadIdx.x;
    float acc = bo[j];
    for (int c = 0; c < 256; ++c) {
        const int pc = (c & (D_ - 1)) * 4 + (c >> 6);
        acc += bc[c] * s[(size_t)b * 256 + c] * Wo[(size_t)pc * COUT_ + j];
    }
    bf[(size_t)b * 256 + j] = acc;
}

// ---------------------------------------------------------------------------
// K8: Wfused[b] = Wc @ Meff[b],   Meff[k][j] = s[b,k]*Wo[(k%64)*4+k/64][j]
// Output written DIRECTLY in packed B-fragment bf16 order for the final GEMM.
// grid (mt=16, b=8), 256 threads = 8 waves, 2 nt tiles per wave.
// ---------------------------------------------------------------------------
__global__ __launch_bounds__(256) void fuse_gemm_kernel(
        const unsigned* __restrict__ WcPk,   // bf16 pairs, row-major [256][128dw]
        const float* __restrict__ Wo, const float* __restrict__ s,
        unsigned* __restrict__ WfPk) {
    const int mt = blockIdx.x;               // CIN tile (M of Wfused)
    const int b  = blockIdx.y;
    const int tid = threadIdx.x, lane = tid & 31, wv = tid >> 5;

    __shared__ float sS[256];
    __shared__ float ldsT[8][2][16][17];     // per-wave D-tile transpose staging

    if (tid < 256) sS[tid] = s[(size_t)b * 256 + tid];
    __syncthreads();

    v8f acc[2] = {vzero8(), vzero8()};

    for (int kt = 0; kt < 8; ++kt) {
        Frag32 a;
        {   // A fragment from Wc (row-major bf16 pairs)
            const unsigned* ap = WcPk + (size_t)(mt * 16 + (lane & 15)) * 128
                                      + kt * 16 + (lane >> 4) * 8;
            a.q[0] = *(const uint4v*)(ap);
            a.q[1] = *(const uint4v*)(ap + 4);
        }
#pragma unroll
        for (int t = 0; t < 2; ++t) {
            const int nt = wv * 2 + t;
            const int n  = nt * 16 + (lane & 15);
            const int kb = kt * 32 + ((lane >> 4) << 4);
            Frag32 bfr;
#pragma unroll
            for (int v = 0; v < 8; ++v) {
                const int k0 = kb + 2 * v, k1 = k0 + 1;
                const float w0 = Wo[(size_t)((k0 & 63) * 4 + (k0 >> 6)) * COUT_ + n] * sS[k0];
                const float w1 = Wo[(size_t)((k1 & 63) * 4 + (k1 >> 6)) * COUT_ + n] * sS[k1];
                bfr.u[v] = f2bf_pk(w0, w1);
            }
            acc[t] = wmma_bf16(a, bfr, acc[t]);
        }
    }

    // D (f32 frag) -> LDS -> packed bf16-fragment global layout
    const int n = lane & 15, hi = lane >> 4;
#pragma unroll
    for (int t = 0; t < 2; ++t)
#pragma unroll
        for (int v = 0; v < 8; ++v) ldsT[wv][t][v + 8 * hi][n] = acc[t][v];
    __syncthreads();

    if ((lane >> 4) == (mt & 1)) {           // this tile fills one K-half
#pragma unroll
        for (int t = 0; t < 2; ++t) {
            const int nt = wv * 2 + t;
            unsigned* dst = WfPk + (size_t)b * 32768
                          + (((size_t)(mt >> 1) * 16 + nt) * 32 + lane) * 8;
#pragma unroll
            for (int v = 0; v < 8; ++v)
                dst[v] = f2bf_pk(ldsT[wv][t][2 * v][n], ldsT[wv][t][2 * v + 1][n]);
        }
    }
}

// ---------------------------------------------------------------------------
// K9: out[b] = x[b] @ Wfused[b] + bfused[b]   (M = B*L, K = 256, N = 256)
// grid 2048 blocks (64 rows), 256 threads = 8 waves:
// mslice = wv&3 (16 rows), nhalf = wv>>2 (128 cols -> 8 tiles).
// 8 B fragments preloaded per K-step; double-buffered LDS A staging (one
// barrier per K-step); coalesced epilogue via LDS transpose.
// ---------------------------------------------------------------------------
__global__ __launch_bounds__(256) void final_gemm_kernel(
        const float* __restrict__ x, const unsigned* __restrict__ WfPk,
        const float* __restrict__ bfused, float* __restrict__ out) {
    __shared__ SMem smem;
    const int tid = threadIdx.x;
    const int lane = tid & 31;
    const int wv = tid >> 5;
    const int mslice = wv & 3;
    const int nhalf = wv >> 2;
    const long gr0 = (long)blockIdx.x * 64;          // row in [0, B*L)
    const int b = (int)(gr0 >> 14);                  // L = 16384
    const unsigned* WB = WfPk + (size_t)b * 32768;

    v8f acc[8];
#pragma unroll
    for (int i = 0; i < 8; ++i) acc[i] = vzero8();

    for (int kt = 0; kt < 8; ++kt) {
        unsigned* buf = smem.a[kt & 1];
        {   // stage raw x tile 64x32 as bf16
            const int r  = tid >> 2;
            const int c8 = (tid & 3) * 8;
            const float* src = x + (gr0 + r) * (long)CIN_ + kt * 32 + c8;
            float4v a0 = *(const float4v*)(src);
            float4v a1 = *(const float4v*)(src + 4);
            unsigned* dst = &buf[r * 20 + (c8 >> 1)];
            dst[0] = f2bf_pk(a0.x, a0.y); dst[1] = f2bf_pk(a0.z, a0.w);
            dst[2] = f2bf_pk(a1.x, a1.y); dst[3] = f2bf_pk(a1.z, a1.w);
            if (kt < 7) __builtin_prefetch(src + 32, 0, 3);   // next K tile
        }
        __syncthreads();

        Frag32 a;
        {
            const int m  = mslice * 16 + (lane & 15);
            const int cd = (lane >> 4) * 8;
            a.q[0] = *(const uint4v*)&buf[m * 20 + cd];
            a.q[1] = *(const uint4v*)&buf[m * 20 + cd + 4];
        }
        const uint4v* bp = (const uint4v*)WB
                         + ((size_t)(kt * 16 + nhalf * 8) * 32 + lane) * 2;
        Frag32 bfr[8];
#pragma unroll
        for (int t = 0; t < 8; ++t) {                // 16 clustered b128 loads
            bfr[t].q[0] = bp[0];
            bfr[t].q[1] = bp[1];
            bp += 64;
        }
#pragma unroll
        for (int t = 0; t < 8; ++t) acc[t] = wmma_bf16(a, bfr[t], acc[t]);
    }

    // epilogue: bias + coalesced b128 stores via per-wave LDS transpose
    __syncthreads();                                  // staging buffers retired
    const float* bias = bfused + (size_t)b * 256;
    const long rbase = gr0 + mslice * 16;
#pragma unroll
    for (int t = 0; t < 8; ++t)
        tile_store(out, rbase, (nhalf * 8 + t) * 16, bias, acc[t], smem.ep[wv], lane);
}

// ---------------------------------------------------------------------------
// Host side
// ---------------------------------------------------------------------------
extern "C" void kernel_launch(void* const* d_in, const int* in_sizes, int n_in,
                              void* d_out, int out_size, void* d_ws, size_t ws_size,
                              hipStream_t stream) {
    const float* x  = (const float*)d_in[0];
    const float* Wa = (const float*)d_in[1];
    const float* ba = (const float*)d_in[2];
    const float* Wb = (const float*)d_in[3];
    const float* bb = (const float*)d_in[4];
    const float* Wc = (const float*)d_in[5];
    const float* bc = (const float*)d_in[6];
    const float* Wo = (const float*)d_in[7];
    const float* bo = (const float*)d_in[8];
    float* out = (float*)d_out;

    // workspace carve-up (~66 MiB total)
    char* p = (char*)d_ws;
    float*    xa     = (float*)p;    p += (size_t)B_ * LS_ * COUT_ * 4;   // 32 MiB
    float*    xb     = (float*)p;    p += (size_t)B_ * LS_ * COUT_ * 4;   // 32 MiB
    float*    pmax   = (float*)p;    p += (size_t)B_ * 16 * 256 * 4;
    float*    pZ     = (float*)p;    p += (size_t)B_ * 16 * 256 * 4;
    float*    pW     = (float*)p;    p += (size_t)B_ * 16 * 256 * 4;
    float*    sbc    = (float*)p;    p += (size_t)B_ * 256 * 4;
    float*    bfused = (float*)p;    p += (size_t)B_ * 256 * 4;
    unsigned* WaPk   = (unsigned*)p; p += (size_t)CIN_ * COUT_ * 2;       // bf16
    unsigned* WbPk   = (unsigned*)p; p += (size_t)CIN_ * COUT_ * 2;
    unsigned* WcPk   = (unsigned*)p; p += (size_t)CIN_ * COUT_ * 2;
    unsigned* WfPk   = (unsigned*)p; p += (size_t)B_ * CIN_ * COUT_ * 2;

    // 1) weight packing / conversion
    pack_b_kernel<<<16, 256, 0, stream>>>(Wa, WaPk);
    pack_b_kernel<<<16, 256, 0, stream>>>(Wb, WbPk);
    cvt_pairs_kernel<<<(CIN_ * COUT_ / 2 + 255) / 256, 256, 0, stream>>>(
        Wc, WcPk, CIN_ * COUT_ / 2);

    // 2) fused maxpool + dual GEMM -> xa, xb
    pool_dual_gemm<<<(B_ * LS_) / 64, 256, 0, stream>>>(x, WaPk, WbPk, ba, bb, xa, xb);

    // 3) softmax-diff reductions -> s[b,c]
    reduce_max_kernel<<<B_ * 16, 256, 0, stream>>>(xa, pmax);
    reduce_sums_kernel<<<B_ * 16, 256, 0, stream>>>(xa, xb, pmax, pZ, pW);
    reduce_final_kernel<<<B_, 256, 0, stream>>>(pZ, pW, sbc);

    // 4) fused bias and fused weight (per batch), weight in packed frag order
    bfused_kernel<<<B_, 256, 0, stream>>>(bc, Wo, bo, sbc, bfused);
    fuse_gemm_kernel<<<dim3(16, B_), 256, 0, stream>>>(WcPk, Wo, sbc, WfPk);

    // 5) final big GEMM: out = x @ Wfused + bfused
    final_gemm_kernel<<<(B_ * (long)L_) / 64, 256, 0, stream>>>(x, WfPk, bfused, out);
}
